// GCN_14087492731266
// MI455X (gfx1250) — matrix operand order repaired and checked by reading, not compiled, not run on hardware
//
#include <hip/hip_runtime.h>
#include <hip/hip_bf16.h>

#define CH 128
#define NUM_GRAPHS 128
#define OUT_CH 10

typedef float    v2f  __attribute__((ext_vector_type(2)));
typedef float    v8f  __attribute__((ext_vector_type(8)));
typedef _Float16 v16h __attribute__((ext_vector_type(16)));

#if defined(__has_builtin)
#  if __has_builtin(__builtin_amdgcn_wmma_f32_16x16x4_f32)
#    define HAVE_WMMA_F32X4 1
#  endif
#endif

// ---------------------------------------------------------------------------
// GEMM: C[nrows,128] = A[nrows,128] @ W[128,128]   (row-major, fp32 in/out)
// block = 256 threads = 8 wave32; wave w -> 16x16 tile (rows blk*16, cols w*16)
// ---------------------------------------------------------------------------
__global__ __launch_bounds__(256) void gcn_gemm128(
    const float* __restrict__ A, const float* __restrict__ W,
    float* __restrict__ C, int nrows) {
  __shared__ float As[16][132];                       // pitch 132 -> no bank conflicts
  const int row0 = blockIdx.x * 16;
  for (int i = threadIdx.x; i < 16 * 128; i += 256) {
    const int r = i >> 7, c = i & 127;
    const int gr = row0 + r;
    As[r][c] = (gr < nrows) ? A[(size_t)gr * CH + c] : 0.0f;
  }
  __syncthreads();

  const int wave = threadIdx.x >> 5;
  const int lane = threadIdx.x & 31;
  const int col0 = wave * 16;
  const int m = lane & 15;          // row (A/C) or col (B/C) within tile
  const int g = lane >> 4;          // half-wave select
  v8f acc = {};

#if HAVE_WMMA_F32X4
  // fp32 WMMA 16x16x4: A vgpr0/1 = K {0,1} (lanes 0-15) or {2,3} (lanes 16-31)
  const int khalf = g * 2;
  for (int k = 0; k < 128; k += 4) {
    v2f a, b;
    a.x = As[m][k + khalf];
    a.y = As[m][k + khalf + 1];
    b.x = W[(size_t)(k + khalf) * CH + col0 + m];
    b.y = W[(size_t)(k + khalf + 1) * CH + col0 + m];
    acc = __builtin_amdgcn_wmma_f32_16x16x4_f32(false, a, false, b,
                                                (short)0, acc, false, false);
  }
#else
  // fallback: fp16 WMMA 16x16x32 (codegen-confirmed), fp32 accumulate
  for (int k0 = 0; k0 < 128; k0 += 32) {
    v16h a, b;
#pragma unroll
    for (int v = 0; v < 8; ++v) {
#pragma unroll
      for (int p = 0; p < 2; ++p) {
        const int ka = g * 16 + v * 2 + p;
        a[v * 2 + p] = (_Float16)As[m][k0 + ka];
        const int kb = g * 16 + v * 2 + p;
        b[v * 2 + p] = (_Float16)W[(size_t)(k0 + kb) * CH + col0 + m];
      }
    }
    acc = __builtin_amdgcn_wmma_f32_16x16x32_f16(false, a, false, b,
                                                 (short)0, acc, false, false);
  }
#endif

  // C/D layout: vgpr r -> M = r + 8*g, N = lane&15
  const int rbase = g * 8;
#pragma unroll
  for (int r = 0; r < 8; ++r) {
    const int gr = row0 + rbase + r;
    if (gr < nrows) C[(size_t)gr * CH + col0 + m] = acc[r];
  }
}

// ---------------------------------------------------------------------------
// small elementwise / scatter kernels
// ---------------------------------------------------------------------------
__global__ void k_fill(float* __restrict__ p, float v, int n) {
  int i = blockIdx.x * 256 + threadIdx.x;
  if (i < n) p[i] = v;
}

__global__ void k_deg_accum(const int* __restrict__ dst, float* __restrict__ deg, int e) {
  int i = blockIdx.x * 256 + threadIdx.x;
  if (i < e) atomicAdd(&deg[dst[i]], 1.0f);
}

__global__ void k_rsqrt_inplace(float* __restrict__ p, int n) {
  int i = blockIdx.x * 256 + threadIdx.x;
  if (i < n) p[i] = rsqrtf(p[i]);          // deg >= 1 (self loop), no zero case
}

// agg[i,c] = t[i,c] * dinv[i]^2 + bias[c]   (self-loop message + bias)
__global__ void k_init_agg(const float* __restrict__ t, const float* __restrict__ dinv,
                           const float* __restrict__ bias, float* __restrict__ agg,
                           int total) {
  int i = blockIdx.x * 256 + threadIdx.x;
  if (i < total) {
    const int node = i >> 7;
    const int c = i & 127;
    const float dv = dinv[node];
    agg[i] = t[i] * dv * dv + bias[c];
  }
}

// one wave32 per edge: lane handles 4 channels (float4 load + 4 f32 atomics)
__global__ __launch_bounds__(256) void k_edge_scatter(
    const float* __restrict__ h, const int* __restrict__ src,
    const int* __restrict__ dst, const float* __restrict__ dinv,
    float* __restrict__ agg, int e) {
  const int warp = (blockIdx.x * 256 + threadIdx.x) >> 5;
  const int lane = threadIdx.x & 31;
  if (warp >= e) return;
  const int s = src[warp];
  const int d = dst[warp];
  const float nrm = dinv[s] * dinv[d];
  const float4 v = ((const float4*)(h + (size_t)s * CH))[lane];
  float* ap = agg + (size_t)d * CH + lane * 4;
  atomicAdd(ap + 0, v.x * nrm);
  atomicAdd(ap + 1, v.y * nrm);
  atomicAdd(ap + 2, v.z * nrm);
  atomicAdd(ap + 3, v.w * nrm);
}

__global__ void k_relu(float* __restrict__ p, int n) {
  int i = blockIdx.x * 256 + threadIdx.x;
  if (i < n) p[i] = fmaxf(p[i], 0.0f);
}

// one wave32 per node: accumulate node row into its graph's pooled row
__global__ __launch_bounds__(256) void k_pool_accum(
    const float* __restrict__ h, const int* __restrict__ batch,
    float* __restrict__ pooled, float* __restrict__ cnt, int n) {
  const int warp = (blockIdx.x * 256 + threadIdx.x) >> 5;
  const int lane = threadIdx.x & 31;
  if (warp >= n) return;
  const int gidx = batch[warp];
  const float4 v = ((const float4*)(h + (size_t)warp * CH))[lane];
  float* pp = pooled + (size_t)gidx * CH + lane * 4;
  atomicAdd(pp + 0, v.x);
  atomicAdd(pp + 1, v.y);
  atomicAdd(pp + 2, v.z);
  atomicAdd(pp + 3, v.w);
  if (lane == 0) atomicAdd(&cnt[gidx], 1.0f);
}

// out[g,o] = (sum_k pooled[g,k]*Wc[k,o]) / max(cnt[g],1) + bc[o]
__global__ void k_classify(const float* __restrict__ pooled, const float* __restrict__ cnt,
                           const float* __restrict__ Wc, const float* __restrict__ bc,
                           float* __restrict__ out) {
  int t = blockIdx.x * 256 + threadIdx.x;
  if (t >= NUM_GRAPHS * OUT_CH) return;
  const int gidx = t / OUT_CH;
  const int o = t % OUT_CH;
  const float c = fmaxf(cnt[gidx], 1.0f);
  float s = 0.0f;
  const float* pr = pooled + (size_t)gidx * CH;
#pragma unroll 8
  for (int k = 0; k < CH; ++k) s += pr[k] * Wc[k * OUT_CH + o];
  out[t] = s / c + bc[o];
}

// ---------------------------------------------------------------------------
extern "C" void kernel_launch(void* const* d_in, const int* in_sizes, int n_in,
                              void* d_out, int out_size, void* d_ws, size_t ws_size,
                              hipStream_t stream) {
  const float* x   = (const float*)d_in[0];
  const int*   ei  = (const int*)d_in[1];   // [2, E] flat: row0 = src, row1 = dst
  const int*   bat = (const int*)d_in[2];
  const float* W1  = (const float*)d_in[3];
  const float* b1  = (const float*)d_in[4];
  const float* W2  = (const float*)d_in[5];
  const float* b2  = (const float*)d_in[6];
  const float* Wc  = (const float*)d_in[7];
  const float* bc  = (const float*)d_in[8];
  float* out = (float*)d_out;

  const int N = in_sizes[0] / CH;
  const int E = in_sizes[1] / 2;
  const int* src = ei;
  const int* dst = ei + E;

  // workspace layout (floats): [bufT | bufA | dinv | pooled | cnt]
  float* ws = (float*)d_ws;
  const size_t NF = (size_t)N * CH;
  float* bufT   = ws;                 // GEMM output (t1 / t2)
  float* bufA   = ws + NF;            // aggregation output (agg1 / agg2)
  float* dinv   = ws + 2 * NF;        // N  (deg -> rsqrt in place)
  float* pooled = dinv + N;           // NUM_GRAPHS*CH
  float* cnt    = pooled + NUM_GRAPHS * CH;  // NUM_GRAPHS

  const int bN   = (N + 255) / 256;
  const int bE   = (E + 255) / 256;
  const int bNF  = (int)((NF + 255) / 256);
  const int bTil = (N + 15) / 16;
  const int bEW  = (E + 7) / 8;       // wave-per-edge, 8 waves/block
  const int bNW  = (N + 7) / 8;       // wave-per-node

  // --- normalization: deg (with self loop) -> dinv
  k_fill<<<bN, 256, 0, stream>>>(dinv, 1.0f, N);
  k_deg_accum<<<bE, 256, 0, stream>>>(dst, dinv, E);
  k_rsqrt_inplace<<<bN, 256, 0, stream>>>(dinv, N);

  // --- layer 1: t1 = x@W1 ; agg1 = scatter(norm * t1) + b1 ; relu
  gcn_gemm128<<<bTil, 256, 0, stream>>>(x, W1, bufT, N);
  k_init_agg<<<bNF, 256, 0, stream>>>(bufT, dinv, b1, bufA, (int)NF);
  k_edge_scatter<<<bEW, 256, 0, stream>>>(bufT, src, dst, dinv, bufA, E);
  k_relu<<<bNF, 256, 0, stream>>>(bufA, (int)NF);

  // --- layer 2: t2 = h1@W2 ; agg2 ; relu   (buffers ping-pong)
  gcn_gemm128<<<bTil, 256, 0, stream>>>(bufA, W2, bufT, N);
  k_init_agg<<<bNF, 256, 0, stream>>>(bufT, dinv, b2, bufA, (int)NF);
  k_edge_scatter<<<bEW, 256, 0, stream>>>(bufT, src, dst, dinv, bufA, E);
  k_relu<<<bNF, 256, 0, stream>>>(bufA, (int)NF);

  // --- mean pool + classifier
  k_fill<<<(NUM_GRAPHS * CH + NUM_GRAPHS + 255) / 256, 256, 0, stream>>>(
      pooled, 0.0f, NUM_GRAPHS * CH + NUM_GRAPHS);
  k_pool_accum<<<bNW, 256, 0, stream>>>(bufA, bat, pooled, cnt, N);
  k_classify<<<(NUM_GRAPHS * OUT_CH + 255) / 256, 256, 0, stream>>>(
      pooled, cnt, Wc, bc, out);
}
